// VSSBlock_44255343018515
// MI455X (gfx1250) — matrix operand order repaired and checked by reading, not compiled, not run on hardware
//
#include <hip/hip_runtime.h>
#include <hip/hip_bf16.h>

typedef __attribute__((ext_vector_type(16))) _Float16 v16h;
typedef __attribute__((ext_vector_type(8)))  float    v8f;

#define DIM     96
#define D_INNER 192
#define D_STATE 16
#define DT_RANK 12
#define BATCH   4
#define HH      56
#define WW      56
#define LSEQ    (HH*WW)          // 3136
#define NROWS   (BATCH*LSEQ)     // 12544
#define PLD     48               // padded SSM-proj row (44 valid)

// ---------------- workspace layout (float offsets) ----------------
#define OFF_XN   0                               // 12544*96      = 1,204,224
#define OFF_XZ   (OFF_XN + NROWS*DIM)            // 12544*384     = 4,816,896
#define OFF_P    (OFF_XZ + NROWS*(2*D_INNER))    // 4*12544*48    = 2,408,448
#define OFF_Y4   (OFF_P  + 4*NROWS*PLD)          // 4*12544*192   = 9,633,792
#define OFF_XRES OFF_P                           // reuse P after scans
#define OFF_X2N  OFF_XN                          // reuse xn after gemm1
#define OFF_HDN  OFF_Y4                          // reuse y4 after combine

// =================== WMMA tile loaders (f32 mem -> f16 regs) ===================
// A-matrix 16x32 (MxK), 16-bit layout per CDNA5 ISA 7.12.2:
//  lanes 0-15: M=lane,   K = e (e<8) / e+8 (e>=8)
//  lanes16-31: M=lane-16, same +8
__device__ inline v16h load_A16x32(const float* A, int lda, int m0, int k0, int lane) {
    int row = m0 + (lane & 15);
    int kb  = k0 + ((lane >> 4) ? 8 : 0);
    const float* p = A + (size_t)row * lda + kb;
    v16h a;
#pragma unroll
    for (int e = 0; e < 8; ++e) a[e]     = (_Float16)p[e];
#pragma unroll
    for (int e = 0; e < 8; ++e) a[8 + e] = (_Float16)p[16 + e];
    return a;
}

// B-matrix 32x16 (KxN) from row-major weight W[N][K]: B(k,n) = W[n0+n][k0+k]
//  lanes 0-15: N=lane, K = e (e<8) / e+8 (e>=8) ; lanes 16-31: N=lane-16, same +8
template<bool NGUARD>
__device__ inline v16h load_B32x16_wT(const float* W, int ldw, int n0, int k0,
                                      int lane, int Nvalid) {
    int col = n0 + (lane & 15);
    v16h b;
    if (!NGUARD || col < Nvalid) {
        const float* p = W + (size_t)col * ldw + k0 + ((lane >> 4) ? 8 : 0);
#pragma unroll
        for (int e = 0; e < 8; ++e) b[e] = (_Float16)p[e];
#pragma unroll
        for (int e = 0; e < 8; ++e) b[8 + e] = (_Float16)p[16 + e];
    } else {
#pragma unroll
        for (int e = 0; e < 16; ++e) b[e] = (_Float16)0.0f;
    }
    return b;
}

// =================== generic WMMA GEMM:  C = A(MxK) * W(NxK)^T ===================
#define EP_STORE    0
#define EP_GELU_B   1   // gelu(acc + bias[col])
#define EP_RES_IN   2   // acc + res[row*ldc+col]
#define EP_RES_B    3   // acc + bias[col] + res[row*ldc+col]

template<int K, int MODE, bool NGUARD>
__global__ __launch_bounds__(128)
void vss_wmma_gemm(const float* __restrict__ A, int lda, long sAz,
                   const float* __restrict__ W, int ldw, long sWz, int Nvalid,
                   float* __restrict__ C, int ldc, long sCz,
                   const float* __restrict__ bias,
                   const float* __restrict__ res) {
    const int lane = threadIdx.x & 31;
    const int wv   = threadIdx.x >> 5;
    const int m0   = (blockIdx.y * 4 + wv) * 16;
    const int n0   = blockIdx.x * 16;
    const int z    = blockIdx.z;

    const float* Az = A + (long)z * sAz;
    const float* Wz = W + (long)z * sWz;
    float*       Cz = C + (long)z * sCz;

    v8f acc;
#pragma unroll
    for (int r = 0; r < 8; ++r) acc[r] = 0.0f;

#pragma unroll
    for (int k0 = 0; k0 < K; k0 += 32) {
        v16h a = load_A16x32(Az, lda, m0, k0, lane);
        v16h b = load_B32x16_wT<NGUARD>(Wz, ldw, n0, k0, lane, Nvalid);
        acc = __builtin_amdgcn_wmma_f32_16x16x32_f16(
                  false, a, false, b, (short)0, acc, false, false);
    }

    // C/D layout: vgpr r -> M = m0 + r + (lane>=16 ? 8 : 0), N = n0 + (lane&15)
    const int col   = n0 + (lane & 15);
    const int rbase = m0 + ((lane >> 4) ? 8 : 0);
#pragma unroll
    for (int r = 0; r < 8; ++r) {
        const int row = rbase + r;
        float v = acc[r];
        if (MODE == EP_GELU_B) {
            v += bias[col];
            v = 0.5f * v * (1.0f + erff(v * 0.70710678118654752f));
        } else if (MODE == EP_RES_IN) {
            v += res[(size_t)row * ldc + col];
        } else if (MODE == EP_RES_B) {
            v += bias[col] + res[(size_t)row * ldc + col];
        }
        Cz[(size_t)row * ldc + col] = v;
    }
}

// =================== LayerNorm over DIM=96 (1 row per wave) ===================
__global__ __launch_bounds__(128)
void vss_layernorm(const float* __restrict__ x, const float* __restrict__ w,
                   const float* __restrict__ b, float* __restrict__ out) {
    const int lane = threadIdx.x & 31;
    const int row  = blockIdx.x * 4 + (threadIdx.x >> 5);
    const float* xr = x + (size_t)row * DIM;
    float v0 = xr[lane], v1 = xr[lane + 32], v2 = xr[lane + 64];
    float s = v0 + v1 + v2;
#pragma unroll
    for (int o = 16; o > 0; o >>= 1) s += __shfl_xor(s, o, 32);
    const float mean = s * (1.0f / 96.0f);
    float d0 = v0 - mean, d1 = v1 - mean, d2 = v2 - mean;
    float q = d0 * d0 + d1 * d1 + d2 * d2;
#pragma unroll
    for (int o = 16; o > 0; o >>= 1) q += __shfl_xor(q, o, 32);
    const float rs = rsqrtf(q * (1.0f / 96.0f) + 1e-6f);
    float* orow = out + (size_t)row * DIM;
    orow[lane]      = d0 * rs * w[lane]      + b[lane];
    orow[lane + 32] = d1 * rs * w[lane + 32] + b[lane + 32];
    orow[lane + 64] = d2 * rs * w[lane + 64] + b[lane + 64];
}

// =================== selective scan: 1 block per (b,dir), 1 lane per channel ===================
__global__ __launch_bounds__(192)
void vss_scan(const float* __restrict__ xz,    // [NROWS][384] ; x_in = cols 0..191
              const float* __restrict__ P,     // [4][NROWS][48] : dt_raw|B|C (at orig pos)
              const float* __restrict__ dtw,   // [4][192][12]
              const float* __restrict__ dtb,   // [4][192]
              const float* __restrict__ Alog,  // [4][192][16]
              const float* __restrict__ Dp,    // [4][192]
              float* __restrict__ y4) {        // [4][NROWS][192]
    const int dir = blockIdx.x & 3;
    const int bb  = blockIdx.x >> 2;
    const int d   = threadIdx.x;

    float dtwv[DT_RANK];
#pragma unroll
    for (int r = 0; r < DT_RANK; ++r) dtwv[r] = dtw[((size_t)dir * D_INNER + d) * DT_RANK + r];
    const float dtbv = dtb[dir * D_INNER + d];
    float Av[D_STATE];
#pragma unroll
    for (int s = 0; s < D_STATE; ++s) Av[s] = -__expf(Alog[((size_t)dir * D_INNER + d) * D_STATE + s]);
    const float Dd = Dp[dir * D_INNER + d];

    float h[D_STATE];
#pragma unroll
    for (int s = 0; s < D_STATE; ++s) h[s] = 0.0f;

    const float* xin = xz + (size_t)bb * LSEQ * (2 * D_INNER);
    const float* Pb  = P  + (size_t)(dir * BATCH + bb) * LSEQ * PLD;
    float*       yo  = y4 + (size_t)(dir * BATCH + bb) * LSEQ * D_INNER;

    for (int t = 0; t < LSEQ; ++t) {
        int idx;
        if      (dir == 0)  idx = t;
        else if (dir == 1)  idx = LSEQ - 1 - t;
        else if (dir == 2)  idx = (t % WW) * WW + t / WW;
        else               { int tp = LSEQ - 1 - t; idx = (tp % WW) * WW + tp / WW; }

        const float* pr = Pb + (size_t)idx * PLD;
        // prefetch next step's P row (hides serial-dependence latency)
        if (t + 1 < LSEQ) {
            int tn = t + 1, in_;
            if      (dir == 0)  in_ = tn;
            else if (dir == 1)  in_ = LSEQ - 1 - tn;
            else if (dir == 2)  in_ = (tn % WW) * WW + tn / WW;
            else               { int tp = LSEQ - 1 - tn; in_ = (tp % WW) * WW + tp / WW; }
            __builtin_prefetch(Pb + (size_t)in_ * PLD, 0, 0);
        }

        float dt = dtbv;
#pragma unroll
        for (int r = 0; r < DT_RANK; ++r) dt += pr[r] * dtwv[r];
        dt = (dt > 20.0f) ? dt : log1pf(__expf(dt));   // softplus

        const float xt = xin[(size_t)idx * (2 * D_INNER) + d];
        float y = xt * Dd;
#pragma unroll
        for (int s = 0; s < D_STATE; ++s) {
            const float bv = pr[DT_RANK + s];
            const float cv = pr[DT_RANK + D_STATE + s];
            h[s] = __expf(dt * Av[s]) * h[s] + dt * bv * xt;
            y += h[s] * cv;
        }
        yo[(size_t)idx * D_INNER + d] = y;
    }
}

// =================== combine: y = (sum_d softmax(dir_w)[d]*y4[d]) * silu(z) ===================
__global__ __launch_bounds__(256)
void vss_combine(const float* __restrict__ y4, const float* __restrict__ dir_w,
                 float* __restrict__ xz) {
    const int i = blockIdx.x * 256 + threadIdx.x;     // i = n*192 + d
    const int n = i / D_INNER;
    const int d = i - n * D_INNER;

    float w0 = dir_w[0], w1 = dir_w[1], w2 = dir_w[2], w3 = dir_w[3];
    float mx = fmaxf(fmaxf(w0, w1), fmaxf(w2, w3));
    float e0 = __expf(w0 - mx), e1 = __expf(w1 - mx), e2 = __expf(w2 - mx), e3 = __expf(w3 - mx);
    const float inv = 1.0f / (e0 + e1 + e2 + e3);

    const long st = (long)NROWS * D_INNER;
    float yv = e0 * y4[i] + e1 * y4[st + i] + e2 * y4[2 * st + i] + e3 * y4[3 * st + i];
    yv *= inv;

    const float z = xz[(size_t)n * (2 * D_INNER) + D_INNER + d];
    const float sil = z / (1.0f + __expf(-z));
    xz[(size_t)n * (2 * D_INNER) + d] = yv * sil;      // overwrite x_in slot with y*silu(z)
}

// =================== launch ===================
extern "C" void kernel_launch(void* const* d_in, const int* in_sizes, int n_in,
                              void* d_out, int out_size, void* d_ws, size_t ws_size,
                              hipStream_t stream) {
    const float* x      = (const float*)d_in[0];
    const float* norm_w = (const float*)d_in[1];
    const float* norm_b = (const float*)d_in[2];
    const float* in_w   = (const float*)d_in[3];
    const float* out_w  = (const float*)d_in[4];
    const float* dtw    = (const float*)d_in[5];
    const float* dtb    = (const float*)d_in[6];
    const float* xw     = (const float*)d_in[7];
    const float* Alog   = (const float*)d_in[8];
    const float* Dp     = (const float*)d_in[9];
    const float* dir_w  = (const float*)d_in[10];
    const float* n2w    = (const float*)d_in[11];
    const float* n2b    = (const float*)d_in[12];
    const float* fw1    = (const float*)d_in[13];
    const float* fb1    = (const float*)d_in[14];
    const float* fw2    = (const float*)d_in[15];
    const float* fb2    = (const float*)d_in[16];
    float* out = (float*)d_out;
    float* ws  = (float*)d_ws;

    float* xn   = ws + OFF_XN;
    float* xz   = ws + OFF_XZ;
    float* P    = ws + OFF_P;
    float* y4   = ws + OFF_Y4;
    float* xres = ws + OFF_XRES;
    float* x2n  = ws + OFF_X2N;
    float* hdn  = ws + OFF_HDN;

    const dim3 blk128(128), blk192(192), blk256(256);
    const int MT4 = NROWS / 64;   // 196 blocks of 4 M-tiles

    // 1) LN1
    vss_layernorm<<<NROWS / 4, blk128, 0, stream>>>(x, norm_w, norm_b, xn);

    // 2) xz = xn * in_w^T   (12544x96 * 96x384)
    vss_wmma_gemm<DIM, EP_STORE, false><<<dim3(24, MT4, 1), blk128, 0, stream>>>(
        xn, DIM, 0, in_w, DIM, 0, 2 * D_INNER, xz, 2 * D_INNER, 0,
        nullptr, nullptr);

    // 3) P[dir] = x_in * xw[dir]^T  (12544x192 * 192x44, padded to 48), 4 dirs
    vss_wmma_gemm<D_INNER, EP_STORE, true><<<dim3(3, MT4, 4), blk128, 0, stream>>>(
        xz, 2 * D_INNER, 0, xw, D_INNER, (long)(DT_RANK + 2 * D_STATE) * D_INNER,
        DT_RANK + 2 * D_STATE, P, PLD, (long)NROWS * PLD,
        nullptr, nullptr);

    // 4) selective scans (16 independent recurrences)
    vss_scan<<<16, blk192, 0, stream>>>(xz, P, dtw, dtb, Alog, Dp, y4);

    // 5) combine directions + silu gate, in place into x_in slots of xz
    vss_combine<<<(NROWS * D_INNER) / 256, blk256, 0, stream>>>(y4, dir_w, xz);

    // 6) xres = shortcut + y * out_w^T  (12544x192 * 192x96)
    vss_wmma_gemm<D_INNER, EP_RES_IN, false><<<dim3(6, MT4, 1), blk128, 0, stream>>>(
        xz, 2 * D_INNER, 0, out_w, D_INNER, 0, DIM, xres, DIM, 0,
        nullptr, x);

    // 7) LN2
    vss_layernorm<<<NROWS / 4, blk128, 0, stream>>>(xres, n2w, n2b, x2n);

    // 8) hdn = gelu(x2n * fw1^T + fb1)  (96 -> 384)
    vss_wmma_gemm<DIM, EP_GELU_B, false><<<dim3(24, MT4, 1), blk128, 0, stream>>>(
        x2n, DIM, 0, fw1, DIM, 0, 4 * DIM, hdn, 4 * DIM, 0,
        fb1, nullptr);

    // 9) out = xres + hdn * fw2^T + fb2  (384 -> 96)
    vss_wmma_gemm<4 * DIM, EP_RES_B, false><<<dim3(6, MT4, 1), blk128, 0, stream>>>(
        hdn, 4 * DIM, 0, fw2, 4 * DIM, 0, DIM, out, DIM, 0,
        fb2, xres);
}
